// MultiHeadSelfAttention_69441031242417
// MI455X (gfx1250) — matrix operand order repaired and checked by reading, not compiled, not run on hardware
//
#include <hip/hip_runtime.h>

typedef __attribute__((ext_vector_type(16))) _Float16 v16h;
typedef __attribute__((ext_vector_type(8)))  _Float16 v8h;
typedef __attribute__((ext_vector_type(4)))  _Float16 v4h;
typedef __attribute__((ext_vector_type(8)))  float    v8f;

#define DEV static __device__ __forceinline__

union F16Frag { uint4 q[2]; v16h h; };

// A-matrix 16x32 f16 fragment (per CDNA5 ISA layout):
// lane L: row = L&15; halves 0..7 -> K = base+0..7, halves 8..15 -> K = 16+base+0..7, base = 8*(L>=16)
DEV v16h load_a_frag(const _Float16* p, int hi) {
  F16Frag f;
  f.q[0] = *(const uint4*)(p + hi * 8);
  f.q[1] = *(const uint4*)(p + hi * 8 + 16);
  return f.h;
}
// B-matrix 32x16 f16 fragment from K-contiguous ("Bt") storage:
// lane L: col = L&15; halves 0..7 -> K = base+0..7, halves 8..15 -> K = 16+base+0..7, base = 8*(L>=16)
DEV v16h load_b_frag(const _Float16* p, int hi) {
  F16Frag f;
  f.q[0] = *(const uint4*)(p + hi * 8);
  f.q[1] = *(const uint4*)(p + hi * 8 + 16);
  return f.h;
}

// CDNA5 async memory->LDS copy (ASYNCcnt-tracked), 16B per lane.
DEV unsigned lds_off(const void* p) { return (unsigned)(uintptr_t)p; }  // low 32b = LDS offset
DEV void async_b128(unsigned ldsaddr, const void* gaddr) {
  asm volatile("global_load_async_to_lds_b128 %0, %1, off"
               :: "v"(ldsaddr), "v"(gaddr) : "memory");
}
DEV void wait_async0() { asm volatile("s_wait_asynccnt 0x0" ::: "memory"); }

// ---------------- conversion kernels ----------------

__global__ __launch_bounds__(256) void cvt_f32_f16(const float* __restrict__ in,
                                                   _Float16* __restrict__ out, int n4) {
  int i = blockIdx.x * 256 + threadIdx.x;
  if (i < n4) {
    float4 v = ((const float4*)in)[i];
    v4h h;
    h[0] = (_Float16)v.x; h[1] = (_Float16)v.y; h[2] = (_Float16)v.z; h[3] = (_Float16)v.w;
    ((v4h*)out)[i] = h;
  }
}

// in[K][N] f32 -> out[N][K] f16
__global__ __launch_bounds__(256) void transpose_cvt(const float* __restrict__ in,
                                                     _Float16* __restrict__ out, int K, int N) {
  int t = blockIdx.x * 256 + threadIdx.x;
  int kc = (t % (K / 8)) * 8;
  int n  = t / (K / 8);
  if (n < N) {
    v8h h;
#pragma unroll
    for (int i = 0; i < 8; i++) h[i] = (_Float16)in[(size_t)(kc + i) * N + n];
    *(v8h*)(&out[(size_t)n * K + kc]) = h;
  }
}

// ---------------- tiled WMMA GEMM: C[M,N] = A[M,K] * Bt[N,K]^T + bias ----------------
// Double-buffered async memory->LDS staging.
// MODE 0: scatter into per-head q (scaled), k, vT (transposed) f16 buffers
// MODE 1: fp32 output

template <int MODE>
__global__ __launch_bounds__(256)
void gemm_kernel(const _Float16* __restrict__ A, const _Float16* __restrict__ Bt,
                 const float* __restrict__ bias,
                 _Float16* __restrict__ qout, _Float16* __restrict__ kout,
                 _Float16* __restrict__ vtout, float* __restrict__ fout,
                 int M, int N, int K) {
  constexpr int BM = 128, BN = 128, BK = 64, LP = 72;  // pitch 72 halves = 144B (16B aligned)
  __shared__ _Float16 sA[2][BM * LP];
  __shared__ _Float16 sB[2][BN * LP];

  const int tid = threadIdx.x;
  const int lane = tid & 31, w = tid >> 5;
  const int row16 = lane & 15, hi = lane >> 4;
  const int wm = w & 3, wn = w >> 2;  // 4 waves in M, 2 in N
  const int m0 = blockIdx.y * BM;
  const int n0 = blockIdx.x * BN;

  auto stage = [&](int k0, int bufi) {
#pragma unroll
    for (int i = 0; i < 4; i++) {
      int c = tid + i * 256;                  // 1024 chunks of 8 halves per tile
      int r = c >> 3, cc = (c & 7) * 8;
      async_b128(lds_off(&sA[bufi][r * LP + cc]), &A[(size_t)(m0 + r) * K + k0 + cc]);
      async_b128(lds_off(&sB[bufi][r * LP + cc]), &Bt[(size_t)(n0 + r) * K + k0 + cc]);
    }
  };

  v8f zero = {};
  v8f acc[2][4];
#pragma unroll
  for (int i = 0; i < 2; i++)
#pragma unroll
    for (int j = 0; j < 4; j++) acc[i][j] = zero;

  stage(0, 0);
  const int T = K / BK;
  for (int t = 0; t < T; t++) {
    wait_async0();       // own async ops done
    __syncthreads();     // everyone's done -> buffer t&1 readable, buffer (t+1)&1 free
    if (t + 1 < T) stage((t + 1) * BK, (t + 1) & 1);
    const _Float16* cA = sA[t & 1];
    const _Float16* cB = sB[t & 1];
#pragma unroll
    for (int ks = 0; ks < BK; ks += 32) {
      v16h af[2], bf[4];
#pragma unroll
      for (int mt = 0; mt < 2; mt++)
        af[mt] = load_a_frag(&cA[(wm * 32 + mt * 16 + row16) * LP + ks], hi);
#pragma unroll
      for (int nt = 0; nt < 4; nt++)
        bf[nt] = load_b_frag(&cB[(wn * 64 + nt * 16 + row16) * LP + ks], hi);
#pragma unroll
      for (int mt = 0; mt < 2; mt++)
#pragma unroll
        for (int nt = 0; nt < 4; nt++)
          acc[mt][nt] = __builtin_amdgcn_wmma_f32_16x16x32_f16(
              false, af[mt], false, bf[nt], (short)0, acc[mt][nt], false, false);
    }
  }

#pragma unroll
  for (int nt = 0; nt < 4; nt++) {
    const int gn = n0 + wn * 64 + nt * 16 + row16;
    const float bv = bias[gn];
#pragma unroll
    for (int mt = 0; mt < 2; mt++) {
#pragma unroll
      for (int r = 0; r < 8; r++) {
        const int gr = m0 + wm * 32 + mt * 16 + r + hi * 8;
        float val = acc[mt][nt][r] + bv;
        if constexpr (MODE == 0) {
          const int head = gn / 192;
          const int sel  = (gn / 64) % 3;
          const int d    = gn & 63;
          const int b    = gr >> 11;
          const int s    = gr & 2047;
          if (sel == 0)
            qout[((size_t)(b * 16 + head) * 2048 + s) * 64 + d] = (_Float16)(val * 0.03125f);
          else if (sel == 1)
            kout[((size_t)(b * 16 + head) * 2048 + s) * 64 + d] = (_Float16)val;
          else
            vtout[((size_t)(b * 16 + head) * 64 + d) * 2048 + s] = (_Float16)val;
        } else {
          fout[(size_t)gr * N + gn] = val;
        }
      }
    }
  }
}

// ---------------- flash attention: per (b,h), 128 query rows per block ----------------

__global__ __launch_bounds__(256)
void attn_kernel(const _Float16* __restrict__ q, const _Float16* __restrict__ k,
                 const _Float16* __restrict__ vT, _Float16* __restrict__ ctx) {
  constexpr int LP = 72;
  __shared__ _Float16 sK[2][64 * LP];
  __shared__ _Float16 sV[2][64 * LP];
  __shared__ _Float16 sP[8 * 16 * LP];  // per-wave P scratch

  const int tid = threadIdx.x, lane = tid & 31, w = tid >> 5;
  const int row16 = lane & 15, hi = lane >> 4;
  const int bh = blockIdx.y;            // b*16 + h
  const int qBase = blockIdx.x * 128;
  const _Float16* qp = q  + (size_t)bh * 2048 * 64;
  const _Float16* kp = k  + (size_t)bh * 2048 * 64;
  const _Float16* vp = vT + (size_t)bh * 64 * 2048;
  _Float16* sPw = &sP[w * 16 * LP];

  auto stageKV = [&](int kb, int bufi) {
#pragma unroll
    for (int i = 0; i < 2; i++) {
      int c = tid + i * 256;              // 512 chunks per tile
      int r = c >> 3, cc = (c & 7) * 8;
      async_b128(lds_off(&sK[bufi][r * LP + cc]), &kp[(size_t)(kb + r) * 64 + cc]);
      async_b128(lds_off(&sV[bufi][r * LP + cc]), &vp[(size_t)r * 2048 + kb + cc]);
    }
  };

  // Q fragments held in registers (scale pre-folded into q buffer)
  const int qrow = qBase + w * 16 + row16;
  v16h qa[2];
#pragma unroll
  for (int f = 0; f < 2; f++) {
    F16Frag fr;
    fr.q[0] = *(const uint4*)(qp + (size_t)qrow * 64 + f * 32 + hi * 8);
    fr.q[1] = *(const uint4*)(qp + (size_t)qrow * 64 + f * 32 + hi * 8 + 16);
    qa[f] = fr.h;
  }

  v8f zero = {};
  v8f o[4];
#pragma unroll
  for (int i = 0; i < 4; i++) o[i] = zero;
  float ml[8], ll[8];
#pragma unroll
  for (int r = 0; r < 8; r++) { ml[r] = -__builtin_inff(); ll[r] = 0.f; }

  stageKV(0, 0);
  const int T = (qBase + 128) / 64;  // causal: keys <= qBase+127
  for (int t = 0; t < T; t++) {
    wait_async0();
    __syncthreads();
    if (t + 1 < T) stageKV((t + 1) * 64, (t + 1) & 1);
    const int kb = t * 64;
    const _Float16* cK = sK[t & 1];
    const _Float16* cV = sV[t & 1];

    if (kb <= qBase + w * 16 + 15) {  // wave-uniform causal skip
      // scores S = Q K^T  (hd = 64 -> 2 k-steps of 32)
      v8f sc[4];
#pragma unroll
      for (int nt = 0; nt < 4; nt++) {
        v8f s = zero;
#pragma unroll
        for (int f = 0; f < 2; f++) {
          v16h kf = load_b_frag(&cK[(nt * 16 + row16) * LP + f * 32], hi);
          s = __builtin_amdgcn_wmma_f32_16x16x32_f16(false, qa[f], false, kf, (short)0, s,
                                                     false, false);
        }
        sc[nt] = s;
      }
      // causal mask (only near the diagonal)
      if (kb + 63 > qBase + w * 16) {
#pragma unroll
        for (int nt = 0; nt < 4; nt++) {
          const int gk = kb + nt * 16 + row16;
#pragma unroll
          for (int r = 0; r < 8; r++) {
            const int qr = qBase + w * 16 + r + hi * 8;
            if (gk > qr) sc[nt][r] = -__builtin_inff();
          }
        }
      }
      // online softmax; each row's 16 cols live across the 16-lane half-group
      float alpha[8];
#pragma unroll
      for (int r = 0; r < 8; r++) {
        float mx = sc[0][r];
#pragma unroll
        for (int nt = 1; nt < 4; nt++) mx = fmaxf(mx, sc[nt][r]);
#pragma unroll
        for (int off = 1; off < 16; off <<= 1) mx = fmaxf(mx, __shfl_xor(mx, off, 32));
        const float mnew = fmaxf(ml[r], mx);
        alpha[r] = __expf(ml[r] - mnew);
        ml[r] = mnew;
        float rs = 0.f;
#pragma unroll
        for (int nt = 0; nt < 4; nt++) {
          const float p = __expf(sc[nt][r] - mnew);
          sc[nt][r] = p;
          rs += p;
        }
#pragma unroll
        for (int off = 1; off < 16; off <<= 1) rs += __shfl_xor(rs, off, 32);
        ll[r] = ll[r] * alpha[r] + rs;
      }
#pragma unroll
      for (int dt = 0; dt < 4; dt++)
#pragma unroll
        for (int r = 0; r < 8; r++) o[dt][r] *= alpha[r];

      // re-layout P (C/D format) -> A format through per-wave LDS (DS in-order per wave)
#pragma unroll
      for (int nt = 0; nt < 4; nt++)
#pragma unroll
        for (int r = 0; r < 8; r++)
          sPw[(r + hi * 8) * LP + nt * 16 + row16] = (_Float16)sc[nt][r];
      v16h pa[2];
#pragma unroll
      for (int f = 0; f < 2; f++) pa[f] = load_a_frag(&sPw[row16 * LP + f * 32], hi);
      // O += P V   (K = 64 keys -> 2 k-steps, N = 64 d -> 4 tiles)
#pragma unroll
      for (int dt = 0; dt < 4; dt++) {
#pragma unroll
        for (int f = 0; f < 2; f++) {
          v16h vf = load_b_frag(&cV[(dt * 16 + row16) * LP + f * 32], hi);
          o[dt] = __builtin_amdgcn_wmma_f32_16x16x32_f16(false, pa[f], false, vf, (short)0,
                                                         o[dt], false, false);
        }
      }
    }
  }

  const int b = bh >> 4, h = bh & 15;
#pragma unroll
  for (int dt = 0; dt < 4; dt++) {
#pragma unroll
    for (int r = 0; r < 8; r++) {
      const int s = qBase + w * 16 + r + hi * 8;
      const int col = h * 64 + dt * 16 + row16;
      ctx[((size_t)(b * 2048 + s)) * 1024 + col] = (_Float16)(o[dt][r] / ll[r]);
    }
  }
}

// ---------------- launch ----------------

extern "C" void kernel_launch(void* const* d_in, const int* in_sizes, int n_in,
                              void* d_out, int out_size, void* d_ws, size_t ws_size,
                              hipStream_t stream) {
  const float* x     = (const float*)d_in[0];
  const float* Wqkv  = (const float*)d_in[1];
  const float* bqkv  = (const float*)d_in[2];
  const float* Wproj = (const float*)d_in[3];
  const float* bproj = (const float*)d_in[4];
  float* y = (float*)d_out;

  char* ws = (char*)d_ws;
  const size_t MB = 1024 * 1024;
  _Float16* xh     = (_Float16*)(ws);             // 16 MB: x in f16 [8192][1024]
  _Float16* wqkvt  = (_Float16*)(ws + 16 * MB);   //  6 MB: Wqkv^T f16 [3072][1024]
  _Float16* wprojt = (_Float16*)(ws + 22 * MB);   //  2 MB: Wproj^T f16 [1024][1024]
  _Float16* qb     = (_Float16*)(ws + 24 * MB);   // 16 MB: q [b][h][s][64] (scaled)
  _Float16* kb     = (_Float16*)(ws + 40 * MB);   // 16 MB: k [b][h][s][64]
  _Float16* vtb    = (_Float16*)(ws + 56 * MB);   // 16 MB: v [b][h][64][s]
  _Float16* ctx    = (_Float16*)(ws + 72 * MB);   // 16 MB: attention out [8192][1024]

  cvt_f32_f16<<<(8192 * 1024 / 4) / 256, 256, 0, stream>>>(x, xh, 8192 * 1024 / 4);
  transpose_cvt<<<(3072 * 1024 / 8) / 256, 256, 0, stream>>>(Wqkv, wqkvt, 1024, 3072);
  transpose_cvt<<<(1024 * 1024 / 8) / 256, 256, 0, stream>>>(Wproj, wprojt, 1024, 1024);

  gemm_kernel<0><<<dim3(3072 / 128, 8192 / 128), 256, 0, stream>>>(
      xh, wqkvt, bqkv, qb, kb, vtb, nullptr, 8192, 3072, 1024);

  attn_kernel<<<dim3(2048 / 128, 64), 256, 0, stream>>>(qb, kb, vtb, ctx);

  gemm_kernel<1><<<dim3(1024 / 128, 8192 / 128), 256, 0, stream>>>(
      ctx, wprojt, bproj, nullptr, nullptr, nullptr, y, 8192, 1024, 1024);
}
